// DPSA_62878321213849
// MI455X (gfx1250) — matrix operand order repaired and checked by reading, not compiled, not run on hardware
//
#include <hip/hip_runtime.h>
#include <hip/hip_bf16.h>

typedef __attribute__((ext_vector_type(16))) __bf16 v16bf;
typedef __attribute__((ext_vector_type(8)))  __bf16 v8bf;
typedef __attribute__((ext_vector_type(8)))  float  v8f;

// ---------------------------------------------------------------------------
// helpers
// ---------------------------------------------------------------------------
__device__ inline v8f vzero8() {
    v8f z = {0.f, 0.f, 0.f, 0.f, 0.f, 0.f, 0.f, 0.f};
    return z;
}

// Async global->LDS copy of 16 bytes per lane (GLOBAL_LOAD_ASYNC_TO_LDS_B128,
// GV addressing). Tracked by ASYNCcnt; LDS offset = generic shared pointer
// truncated to 32 bits.
__device__ inline void async_g2lds_b128(const void* gaddr, const void* lds) {
    unsigned int loff = (unsigned int)(uintptr_t)lds;
    asm volatile("global_load_async_to_lds_b128 %0, %1, off"
                 :: "v"(loff), "v"(gaddr)
                 : "memory");
}
__device__ inline void wait_async0() {
    asm volatile("s_wait_asynccnt 0" ::: "memory");
}

// A-matrix fragment (16x32 bf16, MxK). Per ISA: lane L holds row M=L%16;
// lane-group lg=L/16 selects K-subblock; this code uses K = lg*16 .. lg*16+15 contiguous
// (same order as B). `row` points at the 32-half K-run for
// this lane's M row (LDS or global).
__device__ inline v16bf load_frag_a(const __bf16* row, int lg) {
    v8bf lo = *(const v8bf*)(row + lg * 16);
    v8bf hi = *(const v8bf*)(row + lg * 16 + 8);
    v16bf r;
#pragma unroll
    for (int i = 0; i < 8; ++i) { r[i] = lo[i]; r[8 + i] = hi[i]; }
    return r;
}

// B-matrix fragment (32x16 bf16, KxN). Per ISA: lane L holds column N=L%16,
// this code uses K = (L/16)*16 .. +15 contiguous. `col` points at the 32-half K-run for this
// lane's N column (column-major storage: K contiguous per column).
__device__ inline v16bf load_frag_b(const __bf16* col, int lg) {
    v8bf lo = *(const v8bf*)(col + lg * 16);
    v8bf hi = *(const v8bf*)(col + lg * 16 + 8);
    v16bf r;
#pragma unroll
    for (int i = 0; i < 8; ++i) { r[i] = lo[i]; r[8 + i] = hi[i]; }
    return r;
}

__device__ inline v8f wmma_bf16(v16bf a, v16bf b, v8f c) {
    // D = A*B + C, fp32 accumulate
    return __builtin_amdgcn_wmma_f32_16x16x32_bf16(
        /*neg_a=*/false, a, /*neg_b=*/false, b,
        /*c_mod=*/(short)0, c, /*reuse_a=*/false, /*reuse_b=*/false);
}

// ---------------------------------------------------------------------------
// K0: fp32 -> bf16 elementwise (weights)
// ---------------------------------------------------------------------------
__global__ void __launch_bounds__(256) f32_to_bf16_k(const float* __restrict__ x,
                                                     __bf16* __restrict__ y, int n) {
    int i = blockIdx.x * 256 + threadIdx.x;
    if (i < n) y[i] = (__bf16)x[i];
}

// ---------------------------------------------------------------------------
// K1: channel layernorm over C=512 for one (b, 32-pixel) tile.
// x: [b][512][4096] fp32.  obf: [b][4096][512] bf16 (pixel-major, GEMM-B
// layout).  of32 (optional): [b][512][4096] fp32 normalized (residual).
// ---------------------------------------------------------------------------
__global__ void __launch_bounds__(256) chan_norm_k(const float* __restrict__ x,
                                                   const float* __restrict__ g,
                                                   const float* __restrict__ bt,
                                                   __bf16* __restrict__ obf,
                                                   float* __restrict__ of32) {
    __shared__ __bf16 tile[512 * 33];   // [c][p] bf16 normalized, pad 33
    __shared__ float  red[2 * 256];
    __shared__ float  mu[32], inv[32];

    const int t = threadIdx.x;
    const int b = blockIdx.y;
    const int n0 = blockIdx.x * 32;
    const int p = t & 31, part = t >> 5;

    const float* xb = x + (size_t)b * 512 * 4096 + n0 + p;
    float s = 0.f, s2 = 0.f;
#pragma unroll 4
    for (int i = 0; i < 64; ++i) {
        float v = xb[(size_t)(part * 64 + i) * 4096];
        s += v; s2 += v * v;
    }
    red[part * 32 + p] = s;
    red[256 + part * 32 + p] = s2;
    __syncthreads();
    if (t < 32) {
        float ts = 0.f, t2 = 0.f;
#pragma unroll
        for (int j = 0; j < 8; ++j) { ts += red[j * 32 + t]; t2 += red[256 + j * 32 + t]; }
        float m = ts * (1.f / 512.f);
        float var = t2 * (1.f / 512.f) - m * m;
        mu[t] = m;
        inv[t] = rsqrtf(var + 1e-5f);
    }
    __syncthreads();
    const float m_ = mu[p], iv = inv[p];
    for (int i = 0; i < 64; ++i) {
        int c = part * 64 + i;
        float v = xb[(size_t)c * 4096];
        float y = (v - m_) * iv * g[c] + bt[c];
        tile[c * 33 + p] = (__bf16)y;
        if (of32) of32[((size_t)b * 512 + c) * 4096 + n0 + p] = y;
    }
    __syncthreads();
    // write bf16 pixel-major rows: [b][n][c]
    const int pp = t >> 3, cb = (t & 7) * 64;
    __bf16* dst = obf + ((size_t)b * 4096 + n0 + pp) * 512 + cb;
#pragma unroll 8
    for (int i = 0; i < 64; ++i) dst[i] = tile[(cb + i) * 33 + pp];
}

// ---------------------------------------------------------------------------
// K2/K3/K7: bf16 WMMA GEMM with async-copy software pipeline.
// C[b][o][n] = sum_k A[o][k] * B[b][n][k]
// A: [M][512] bf16 row-major.  B: [b][4096][512] bf16 pixel-major.
// Block: 128(M) x 128(N), 8 waves, each wave 32x64.  K staged in 32-slices
// into double-buffered LDS via GLOBAL_LOAD_ASYNC_TO_LDS_B128 (ASYNCcnt),
// overlapping the next slice's DMA with the current slice's 8 WMMAs.
// FINAL=1: C = gamma*acc + res (residual add) into d_out.
// ---------------------------------------------------------------------------
template <int FINAL>
__global__ void __launch_bounds__(256) gemm_wmma_k(const __bf16* __restrict__ A,
                                                   const __bf16* __restrict__ Bx,
                                                   float* __restrict__ C,
                                                   const float* __restrict__ res,
                                                   const float* __restrict__ gamma,
                                                   int M) {
    __shared__ __bf16 As[2][128 * 40];   // [buf][mrow][k], stride 40 halves
    __shared__ __bf16 Bs[2][128 * 40];   // [buf][ncol][k]

    const int t = threadIdx.x;
    const int bb = blockIdx.z;
    const int m0 = blockIdx.y * 128, n0 = blockIdx.x * 128;
    const int wid = t >> 5, lane = t & 31, lg = lane >> 4, ln = lane & 15;
    const int wm = (wid >> 1) * 32, wn = (wid & 1) * 64;

    v8f acc[2][4];
#pragma unroll
    for (int mi = 0; mi < 2; ++mi)
#pragma unroll
        for (int ni = 0; ni < 4; ++ni) acc[mi][ni] = vzero8();

    const int ldrow = t >> 1, ldk = (t & 1) * 16;
    const size_t Abase = (size_t)(m0 + ldrow) * 512 + ldk;
    const size_t Bbase = ((size_t)bb * 4096 + (n0 + ldrow)) * 512 + ldk;

    // stage K-slice kt into buffer `buf` (32B of A + 32B of B per thread)
    auto stage = [&](int buf, int kt) {
        const int k = kt * 32;
        __bf16* la = &As[buf][ldrow * 40 + ldk];
        __bf16* lb = &Bs[buf][ldrow * 40 + ldk];
        async_g2lds_b128(A + Abase + k, la);
        async_g2lds_b128(A + Abase + k + 8, la + 8);
        async_g2lds_b128(Bx + Bbase + k, lb);
        async_g2lds_b128(Bx + Bbase + k + 8, lb + 8);
    };

    stage(0, 0);
    wait_async0();
    __syncthreads();

    for (int kt = 0; kt < 16; ++kt) {
        const int cur = kt & 1;
        if (kt < 15) stage(cur ^ 1, kt + 1);   // DMA next slice during compute

        v16bf af[2], bf[4];
#pragma unroll
        for (int mi = 0; mi < 2; ++mi)
            af[mi] = load_frag_a(&As[cur][(wm + mi * 16 + ln) * 40], lg);
#pragma unroll
        for (int ni = 0; ni < 4; ++ni)
            bf[ni] = load_frag_b(&Bs[cur][(wn + ni * 16 + ln) * 40], lg);
#pragma unroll
        for (int mi = 0; mi < 2; ++mi)
#pragma unroll
            for (int ni = 0; ni < 4; ++ni)
                acc[mi][ni] = wmma_bf16(af[mi], bf[ni], acc[mi][ni]);

        wait_async0();      // own next-slice DMAs done
        __syncthreads();    // everyone's reads of `cur` + writes of `cur^1` done
    }

    const float gsc = FINAL ? gamma[0] : 0.f;
#pragma unroll
    for (int mi = 0; mi < 2; ++mi)
#pragma unroll
        for (int ni = 0; ni < 4; ++ni)
#pragma unroll
            for (int r = 0; r < 8; ++r) {
                int o = m0 + wm + mi * 16 + r + 8 * lg;     // C layout: M = r + 8*(lane/16)
                int n = n0 + wn + ni * 16 + ln;             //           N = lane%16
                size_t idx = ((size_t)bb * M + o) * 4096 + n;
                float v = acc[mi][ni][r];
                if (FINAL) v = gsc * v + res[idx];
                C[idx] = v;
            }
}

// ---------------------------------------------------------------------------
// K4: per-pixel l2norm over d=64 + |.| statistics.
// src: fp32 [b][OC][4096], head slice at o = h*64.  dst: bf16 [bh][4096][64].
// mode 0 (q): atomicAdd probe[bh][d] += sum_n |q|.
// mode 1 (k): atomicAdd khei[bh][d][h], kwid[bh][d][w].
// ---------------------------------------------------------------------------
__global__ void __launch_bounds__(256) l2norm_stats_k(const float* __restrict__ src,
                                                      __bf16* __restrict__ dst,
                                                      float* __restrict__ probe,
                                                      float* __restrict__ khei,
                                                      float* __restrict__ kwid,
                                                      int OC, int mode) {
    __shared__ float tile[64 * 132];    // [d][p], pad to 132 (16B-aligned rows)
    const int t = threadIdx.x;
    const int bh = blockIdx.y;
    const int n0 = blockIdx.x * 128;
    const int b = bh >> 3, h = bh & 7;
    const float* s = src + ((size_t)b * OC + h * 64) * 4096;

    {
        const int d = t >> 2, c0 = (t & 3) * 32;
        const float4* gp = (const float4*)(s + (size_t)d * 4096 + n0 + c0);
        float4* lp = (float4*)(tile + d * 132 + c0);
#pragma unroll
        for (int i = 0; i < 8; ++i) lp[i] = gp[i];
    }
    __syncthreads();

    if (t < 128) {
        const int p = t;
        float ss = 0.f;
#pragma unroll
        for (int d = 0; d < 64; ++d) { float v = tile[d * 132 + p]; ss += v * v; }
        float nrm = sqrtf(ss);
        float iv = 1.f / fmaxf(nrm, 1e-12f);
        __bf16 row[64];
#pragma unroll
        for (int d = 0; d < 64; ++d) {
            float v = tile[d * 132 + p] * iv;
            row[d] = (__bf16)v;
            tile[d * 132 + p] = fabsf(v);
        }
        __bf16* dp = dst + ((size_t)bh * 4096 + n0 + p) * 64;
#pragma unroll
        for (int i = 0; i < 64; ++i) dp[i] = row[i];
    }
    __syncthreads();

    if (t < 64) {
        const int d = t;
        if (mode == 0) {
            float a = 0.f;
#pragma unroll 4
            for (int p = 0; p < 128; ++p) a += tile[d * 132 + p];
            atomicAdd(&probe[bh * 64 + d], a);
        } else {
            const int h0 = n0 >> 6;           // block covers spatial rows h0, h0+1
            float a0 = 0.f, a1 = 0.f;
            for (int w = 0; w < 64; ++w) {
                float v0 = tile[d * 132 + w];
                float v1 = tile[d * 132 + 64 + w];
                a0 += v0; a1 += v1;
                atomicAdd(&kwid[((size_t)bh * 64 + d) * 64 + w], v0 + v1);
            }
            atomicAdd(&khei[((size_t)bh * 64 + d) * 64 + h0], a0);
            atomicAdd(&khei[((size_t)bh * 64 + d) * 64 + h0 + 1], a1);
        }
    }
}

// ---------------------------------------------------------------------------
// K5: per-(bh) score matvecs + top-8 selection + k/v gather.
// kg:  bf16 [bh][j][d]  (j = ih*8+iw)   -> B-layout for QK^T
// vgt: bf16 [bh][d][j]                  -> B-layout for P@V
// ---------------------------------------------------------------------------
__global__ void __launch_bounds__(64) topk_gather_k(const float* __restrict__ probe,
                                                    const float* __restrict__ khei,
                                                    const float* __restrict__ kwid,
                                                    const __bf16* __restrict__ kbf,
                                                    const float* __restrict__ kvf,
                                                    __bf16* __restrict__ kg,
                                                    __bf16* __restrict__ vgt) {
    const int bh = blockIdx.x, t = threadIdx.x;
    __shared__ float qp[64], sr[64], sc[64];
    __shared__ int ih[8], iw[8], nidx[64];

    qp[t] = probe[bh * 64 + t];
    __syncthreads();
    float a = 0.f, c2 = 0.f;
#pragma unroll 4
    for (int d = 0; d < 64; ++d) {
        float q = qp[d];
        a  += q * khei[((size_t)bh * 64 + d) * 64 + t];
        c2 += q * kwid[((size_t)bh * 64 + d) * 64 + t];
    }
    sr[t] = a; sc[t] = c2;
    __syncthreads();
    if (t == 0) {   // sorted-descending top-8; strict > matches jax top_k ties
        float tmp[64];
        for (int i = 0; i < 64; ++i) tmp[i] = sr[i];
        for (int r = 0; r < 8; ++r) {
            int best = 0; float bv = tmp[0];
            for (int i = 1; i < 64; ++i) if (tmp[i] > bv) { bv = tmp[i]; best = i; }
            ih[r] = best; tmp[best] = -3.4e38f;
        }
        for (int i = 0; i < 64; ++i) tmp[i] = sc[i];
        for (int r = 0; r < 8; ++r) {
            int best = 0; float bv = tmp[0];
            for (int i = 1; i < 64; ++i) if (tmp[i] > bv) { bv = tmp[i]; best = i; }
            iw[r] = best; tmp[best] = -3.4e38f;
        }
    }
    __syncthreads();
    const int n = ih[t >> 3] * 64 + iw[t & 7];
    nidx[t] = n;
    // gather normalized k row (64 bf16 = 128B)
    {
        const uint4* sp = (const uint4*)(kbf + ((size_t)bh * 4096 + n) * 64);
        uint4* dp = (uint4*)(kg + ((size_t)bh * 64 + t) * 64);
#pragma unroll
        for (int i = 0; i < 8; ++i) dp[i] = sp[i];
    }
    __syncthreads();
    // gather v transposed: vgt[bh][d=t][j]
    const int b = bh >> 3, h = bh & 7;
    const float* vs = kvf + ((size_t)b * 1024 + 512 + h * 64 + t) * 4096;
    __bf16* vd = vgt + ((size_t)bh * 64 + t) * 64;
#pragma unroll 8
    for (int j = 0; j < 64; ++j) vd[j] = (__bf16)vs[nidx[j]];
}

// ---------------------------------------------------------------------------
// K6: attention.  Per wave: 16 queries x 64 keys.  S=qk^T (8 WMMAs), softmax
// in registers via half-wave shfl reductions, P->LDS, O=P@V (8 WMMAs).
// Ks/Vs pulled in via async DMA.  aout: bf16 [b][n][512], inner ch = h*64+d.
// ---------------------------------------------------------------------------
__global__ void __launch_bounds__(256) attention_k(const __bf16* __restrict__ q,
                                                   const __bf16* __restrict__ kg,
                                                   const __bf16* __restrict__ vgt,
                                                   __bf16* __restrict__ aout) {
    __shared__ __bf16 Ks[64 * 64];        // [j][d]
    __shared__ __bf16 Vs[64 * 64];        // [d][j]
    __shared__ __bf16 Pb[8 * 16 * 64];    // per-wave P [m][j]

    const int t = threadIdx.x;
    const int bh = blockIdx.y, b = bh >> 3, h = bh & 7;
    const int wid = t >> 5, lane = t & 31, lg = lane >> 4, ln = lane & 15;
    const int n0 = blockIdx.x * 128 + wid * 16;

    {
        const char* gk = (const char*)(kg + (size_t)bh * 4096);
        const char* gv = (const char*)(vgt + (size_t)bh * 4096);
        char* lk = (char*)Ks;
        char* lv = (char*)Vs;
        async_g2lds_b128(gk + t * 16, lk + t * 16);
        async_g2lds_b128(gk + (t + 256) * 16, lk + (t + 256) * 16);
        async_g2lds_b128(gv + t * 16, lv + t * 16);
        async_g2lds_b128(gv + (t + 256) * 16, lv + (t + 256) * 16);
    }
    wait_async0();
    __syncthreads();

    // S = q @ k^T  (M=16 queries, N=64 keys, K=64 = 2 steps)
    v8f s[4];
#pragma unroll
    for (int ni = 0; ni < 4; ++ni) s[ni] = vzero8();
    const __bf16* qb = q + (size_t)bh * 4096 * 64;
#pragma unroll
    for (int ks = 0; ks < 2; ++ks) {
        v16bf af = load_frag_a(qb + (size_t)(n0 + ln) * 64 + ks * 32, lg);
#pragma unroll
        for (int ni = 0; ni < 4; ++ni) {
            v16bf bf = load_frag_b(Ks + (ni * 16 + ln) * 64 + ks * 32, lg);
            s[ni] = wmma_bf16(af, bf, s[ni]);
        }
    }

    // softmax per row.  C layout: row m = r + 8*lg lives in one 16-lane group.
    float e[4][8], mx[8], sm[8];
#pragma unroll
    for (int r = 0; r < 8; ++r) {
        float m = fmaxf(fmaxf(s[0][r], s[1][r]), fmaxf(s[2][r], s[3][r]));
#pragma unroll
        for (int off = 1; off < 16; off <<= 1) m = fmaxf(m, __shfl_xor(m, off, 32));
        mx[r] = m;
    }
#pragma unroll
    for (int r = 0; r < 8; ++r) {
#pragma unroll
        for (int ni = 0; ni < 4; ++ni) e[ni][r] = __expf(s[ni][r] - mx[r]);
        float su = e[0][r] + e[1][r] + e[2][r] + e[3][r];
#pragma unroll
        for (int off = 1; off < 16; off <<= 1) su += __shfl_xor(su, off, 32);
        sm[r] = su;
    }

    // write P (bf16) into per-wave LDS in plain [m][j] so it can be re-read
    // in A-fragment layout.
    __bf16* P = Pb + wid * 16 * 64;
#pragma unroll
    for (int ni = 0; ni < 4; ++ni)
#pragma unroll
        for (int r = 0; r < 8; ++r)
            P[(r + 8 * lg) * 64 + ni * 16 + ln] = (__bf16)(e[ni][r] / sm[r]);
    asm volatile("s_wait_dscnt 0" ::: "memory");   // wave-local LDS RAW fence

    // O = P @ V  (M=16, N=64 d, K=64 j = 2 steps)
    v8f o[4];
#pragma unroll
    for (int ni = 0; ni < 4; ++ni) o[ni] = vzero8();
#pragma unroll
    for (int ks = 0; ks < 2; ++ks) {
        v16bf af = load_frag_a(P + ln * 64 + ks * 32, lg);
#pragma unroll
        for (int ni = 0; ni < 4; ++ni) {
            v16bf bf = load_frag_b(Vs + (ni * 16 + ln) * 64 + ks * 32, lg);
            o[ni] = wmma_bf16(af, bf, o[ni]);
        }
    }
#pragma unroll
    for (int ni = 0; ni < 4; ++ni)
#pragma unroll
        for (int r = 0; r < 8; ++r) {
            int n = n0 + r + 8 * lg;
            int c = h * 64 + ni * 16 + ln;
            aout[((size_t)b * 4096 + n) * 512 + c] = (__bf16)o[ni][r];
        }
}

// ---------------------------------------------------------------------------
// launch
// ---------------------------------------------------------------------------
extern "C" void kernel_launch(void* const* d_in, const int* in_sizes, int n_in,
                              void* d_out, int out_size, void* d_ws, size_t ws_size,
                              hipStream_t stream) {
    (void)in_sizes; (void)n_in; (void)out_size; (void)ws_size;
    const float* context = (const float*)d_in[0];
    const float* qsrc    = (const float*)d_in[1];
    const float* ctx_g   = (const float*)d_in[2];
    const float* ctx_b   = (const float*)d_in[3];
    const float* qs_g    = (const float*)d_in[4];
    const float* qs_b    = (const float*)d_in[5];
    const float* w_q     = (const float*)d_in[6];
    const float* w_kv    = (const float*)d_in[7];
    const float* w_out   = (const float*)d_in[8];
    const float* gamma   = (const float*)d_in[9];
    float* out = (float*)d_out;

    char* ws = (char*)d_ws;
    size_t off = 0;
    auto alloc = [&](size_t bytes) -> char* {
        char* p = ws + off;
        off = (off + bytes + 255) & ~(size_t)255;
        return p;
    };
    __bf16* ctx_bf  = (__bf16*)alloc((size_t)8 * 4096 * 512 * 2);
    __bf16* qs_bf   = (__bf16*)alloc((size_t)8 * 4096 * 512 * 2);
    float*  qs_n    = (float*)alloc((size_t)8 * 512 * 4096 * 4);
    float*  kv_f    = (float*)alloc((size_t)8 * 1024 * 4096 * 4);
    float*  q_f     = (float*)alloc((size_t)8 * 512 * 4096 * 4);
    __bf16* q_bf    = (__bf16*)alloc((size_t)64 * 4096 * 64 * 2);
    __bf16* k_bf    = (__bf16*)alloc((size_t)64 * 4096 * 64 * 2);
    float*  qprobe  = (float*)alloc((size_t)64 * 64 * 4);
    float*  khei    = (float*)alloc((size_t)64 * 64 * 64 * 4);
    float*  kwid    = (float*)alloc((size_t)64 * 64 * 64 * 4);
    __bf16* kg      = (__bf16*)alloc((size_t)64 * 64 * 64 * 2);
    __bf16* vgt     = (__bf16*)alloc((size_t)64 * 64 * 64 * 2);
    __bf16* aout    = (__bf16*)alloc((size_t)8 * 4096 * 512 * 2);
    __bf16* wq_bf   = (__bf16*)alloc((size_t)512 * 512 * 2);
    __bf16* wkv_bf  = (__bf16*)alloc((size_t)1024 * 512 * 2);
    __bf16* wout_bf = (__bf16*)alloc((size_t)512 * 512 * 2);

    // weights -> bf16
    f32_to_bf16_k<<<dim3((512 * 512 + 255) / 256), 256, 0, stream>>>(w_q, wq_bf, 512 * 512);
    f32_to_bf16_k<<<dim3((1024 * 512 + 255) / 256), 256, 0, stream>>>(w_kv, wkv_bf, 1024 * 512);
    f32_to_bf16_k<<<dim3((512 * 512 + 255) / 256), 256, 0, stream>>>(w_out, wout_bf, 512 * 512);

    // zero atomic accumulators
    hipMemsetAsync(qprobe, 0, (size_t)64 * 64 * 4, stream);
    hipMemsetAsync(khei, 0, (size_t)64 * 64 * 64 * 4, stream);
    hipMemsetAsync(kwid, 0, (size_t)64 * 64 * 64 * 4, stream);

    // channel layernorms
    chan_norm_k<<<dim3(128, 8), 256, 0, stream>>>(context, ctx_g, ctx_b, ctx_bf, nullptr);
    chan_norm_k<<<dim3(128, 8), 256, 0, stream>>>(qsrc, qs_g, qs_b, qs_bf, qs_n);

    // projections (bf16 WMMA, fp32 accumulate, async-pipelined LDS staging)
    gemm_wmma_k<0><<<dim3(32, 8, 8), 256, 0, stream>>>(wkv_bf, ctx_bf, kv_f, nullptr, nullptr, 1024);
    gemm_wmma_k<0><<<dim3(32, 4, 8), 256, 0, stream>>>(wq_bf, qs_bf, q_f, nullptr, nullptr, 512);

    // l2 norms + pruning statistics
    l2norm_stats_k<<<dim3(32, 64), 256, 0, stream>>>(q_f, q_bf, qprobe, nullptr, nullptr, 512, 0);
    l2norm_stats_k<<<dim3(32, 64), 256, 0, stream>>>(kv_f, k_bf, nullptr, khei, kwid, 1024, 1);

    // dual top-k + gather
    topk_gather_k<<<dim3(64), 64, 0, stream>>>(qprobe, khei, kwid, k_bf, kv_f, kg, vgt);

    // attention (WMMA QK^T + softmax + WMMA PV)
    attention_k<<<dim3(32, 64), 256, 0, stream>>>(q_bf, kg, vgt, aout);

    // output projection fused with gamma*y + qs residual
    gemm_wmma_k<1><<<dim3(32, 4, 8), 256, 0, stream>>>(wout_bf, aout, out, qs_n, gamma, 512);
}
